// contextual_layers_75505525064739
// MI455X (gfx1250) — compile-verified, matrix operand order checked by reading, not run on hardware
//
#include <hip/hip_runtime.h>
#include <hip/hip_bf16.h>

#define N_NODES 50000
#define N_EDGES 800000
#define DIM 128
#define NEG_SLOPE 0.2f

typedef __attribute__((ext_vector_type(16))) __bf16 v16bf;
typedef __attribute__((ext_vector_type(8)))  __bf16 v8bf;
typedef __attribute__((ext_vector_type(8)))  float  v8f;

// ---------------------------------------------------------------------------
// W (fp32, KxD row-major) -> Wt (bf16, column-major: Wt[col*128 + k]).
// 32 KB result, stays WGP$/L2 resident for the whole GEMM.
// ---------------------------------------------------------------------------
__global__ void gat_wcvt(const float* __restrict__ W, __bf16* __restrict__ Wt)
{
  const int i = blockIdx.x * blockDim.x + threadIdx.x;
  if (i >= DIM * DIM) return;
  const int k   = i >> 7;
  const int col = i & (DIM - 1);
  Wt[(size_t)col * DIM + k] = (__bf16)W[i];
}

// ---------------------------------------------------------------------------
// Fused: z = x @ W (WMMA bf16), el = z.al, er = z.ar.
// One wave per 16-row strip; the wave computes all 8 column tiles:
//   - A fragments (4 K-blocks) loaded once, converted fp32->bf16
//   - B fragments: two contiguous 16B bf16 loads from Wt per tile/K-block
//   - 32x v_wmma_f32_16x16x32_bf16 per wave
// Epilogue: store z, then fold the full 16x128 strip into el/er via per-lane
// partials + xor-shuffle reduction inside each 16-lane half (C layout:
// lanes 0-15 hold rows r, lanes 16-31 rows r+8; together each half's 16 lanes
// span all 128 columns across the 8 tiles).
// ---------------------------------------------------------------------------
__global__ __launch_bounds__(256) void gat_gemm_wmma(
    const float* __restrict__ X, const __bf16* __restrict__ Wt,
    const float* __restrict__ al, const float* __restrict__ ar,
    float* __restrict__ Z, float* __restrict__ el, float* __restrict__ er)
{
  const int lane  = threadIdx.x & 31;
  const int wv    = threadIdx.x >> 5;
  const int strip = blockIdx.x * 8 + wv;
  if (strip >= N_NODES / 16) return;
  const int m    = lane & 15;
  const int half = lane >> 4;          // 0 or 1
  const int row0 = strip * 16;

  const float* __restrict__ xrow = X + (size_t)(row0 + m) * DIM;

  // A fragments for all 4 K-blocks (ISA 7.12.2 16-bit A 16x32 layout)
  v16bf afrag[4];
#pragma unroll
  for (int kb4 = 0; kb4 < 4; ++kb4) {
    const int kb = kb4 * 32;
#pragma unroll
    for (int i = 0; i < 8; ++i) {
      afrag[kb4][i]     = (__bf16)xrow[kb + half * 8 + i];
      afrag[kb4][i + 8] = (__bf16)xrow[kb + 16 + half * 8 + i];
    }
  }

  v8f c[8];
#pragma unroll
  for (int t = 0; t < 8; ++t) c[t] = (v8f){};

#pragma unroll
  for (int kb4 = 0; kb4 < 4; ++kb4) {
    const int kb = kb4 * 32;
#pragma unroll
    for (int t = 0; t < 8; ++t) {
      const __bf16* wp = Wt + (size_t)(t * 16 + m) * DIM + kb + half * 8;
      const v8bf blo = *(const v8bf*)wp;          // K = kb+half*8 .. +7
      const v8bf bhi = *(const v8bf*)(wp + 16);   // K = kb+16+half*8 .. +7
      const v16bf b = __builtin_shufflevector(
          blo, bhi, 0, 1, 2, 3, 4, 5, 6, 7, 8, 9, 10, 11, 12, 13, 14, 15);
      c[t] = __builtin_amdgcn_wmma_f32_16x16x32_bf16(
          false, afrag[kb4], false, b, (short)0, c[t], false, false);
    }
  }

  // Store z (coalesced 64B per (tile,row,half) across the 16 lanes)
#pragma unroll
  for (int t = 0; t < 8; ++t) {
#pragma unroll
    for (int r = 0; r < 8; ++r) {
      Z[(size_t)(row0 + r + half * 8) * DIM + t * 16 + m] = c[t][r];
    }
  }

  // Fused attention logits: partials per row held by this lane
  float pl[8], pr[8];
#pragma unroll
  for (int r = 0; r < 8; ++r) { pl[r] = 0.f; pr[r] = 0.f; }
#pragma unroll
  for (int t = 0; t < 8; ++t) {
    const float alv = al[t * 16 + m];
    const float arv = ar[t * 16 + m];
#pragma unroll
    for (int r = 0; r < 8; ++r) {
      pl[r] += c[t][r] * alv;
      pr[r] += c[t][r] * arv;
    }
  }
  // Reduce across the 16 lanes of this half (xor masks < 16 stay in-half)
#pragma unroll
  for (int off = 8; off > 0; off >>= 1) {
#pragma unroll
    for (int r = 0; r < 8; ++r) {
      pl[r] += __shfl_xor(pl[r], off, 32);
      pr[r] += __shfl_xor(pr[r], off, 32);
    }
  }
  if (m == 0) {
#pragma unroll
    for (int r = 0; r < 8; ++r) {
      el[row0 + r + half * 8] = pl[r];
      er[row0 + r + half * 8] = pr[r];
    }
  }
}

// acc = 0 over N*D; m = -inf, s = 0 over N
__global__ void gat_init(float* __restrict__ acc, float* __restrict__ mmax,
                         float* __restrict__ ssum, int n, int total)
{
  const int i = blockIdx.x * blockDim.x + threadIdx.x;
  if (i < total) acc[i] = 0.f;
  if (i < n) { mmax[i] = -__builtin_inff(); ssum[i] = 0.f; }
}

__device__ __forceinline__ void atomicMaxF(float* addr, float v)
{
  int* ai = (int*)addr;
  int old = __float_as_int(*addr);
  while (__int_as_float(old) < v) {
    const int assumed = old;
    old = atomicCAS(ai, assumed, __float_as_int(v));
    if (old == assumed) break;
  }
}

// e = leaky_relu(el[src]+er[dst]); segment max over dst
__global__ void gat_edge_logit(const int* __restrict__ src,
                               const int* __restrict__ dst,
                               const float* __restrict__ el,
                               const float* __restrict__ er,
                               float* __restrict__ ebuf,
                               float* __restrict__ mmax, int ne)
{
  const int i = blockIdx.x * blockDim.x + threadIdx.x;
  if (i >= ne) return;
  float e = el[src[i]] + er[dst[i]];
  e = (e > 0.f) ? e : NEG_SLOPE * e;
  ebuf[i] = e;
  atomicMaxF(&mmax[dst[i]], e);
}

// ex = exp(e - m[dst]); segment sum over dst
__global__ void gat_edge_exp(const int* __restrict__ dst,
                             float* __restrict__ ebuf,
                             const float* __restrict__ mmax,
                             float* __restrict__ ssum, int ne)
{
  const int i = blockIdx.x * blockDim.x + threadIdx.x;
  if (i >= ne) return;
  const int d = dst[i];
  const float ex = __expf(ebuf[i] - mmax[d]);
  ebuf[i] = ex;
  unsafeAtomicAdd(&ssum[d], ex);
}

// acc[dst] += (ex/s[dst]) * z[src] : one wave per edge, float4 per lane,
// native global_atomic_add_f32 into the L2-resident accumulator.
__global__ __launch_bounds__(256) void gat_aggregate(
    const int* __restrict__ src, const int* __restrict__ dst,
    const float* __restrict__ ebuf, const float* __restrict__ ssum,
    const float* __restrict__ Z, float* __restrict__ acc, int ne)
{
  const int lane = threadIdx.x & 31;
  const int wave = threadIdx.x >> 5;
  const int e = blockIdx.x * 8 + wave;
  if (e >= ne) return;
  const int s = src[e];
  const int d = dst[e];
  const float coeff = ebuf[e] / ssum[d];
  const float4 zv = ((const float4*)(Z + (size_t)s * DIM))[lane];
  float* ap = acc + (size_t)d * DIM + lane * 4;
  unsafeAtomicAdd(ap + 0, coeff * zv.x);
  unsafeAtomicAdd(ap + 1, coeff * zv.y);
  unsafeAtomicAdd(ap + 2, coeff * zv.z);
  unsafeAtomicAdd(ap + 3, coeff * zv.w);
}

// out = relu(acc + b)
__global__ void gat_bias_relu(const float* __restrict__ acc,
                              const float* __restrict__ bias,
                              float* __restrict__ out, int total)
{
  const int i = blockIdx.x * blockDim.x + threadIdx.x;
  if (i >= total) return;
  const float v = acc[i] + bias[i & (DIM - 1)];
  out[i] = v > 0.f ? v : 0.f;
}

// ---------------------------------------------------------------------------
static void run_layer(const float* x, const float* W, const float* al,
                      const float* ar, const float* b,
                      float* zbuf, float* acc, float* hout, __bf16* wtbuf,
                      float* el, float* er, float* mmax, float* ssum,
                      float* ebuf, const int* src, const int* dst,
                      hipStream_t stream)
{
  const dim3 b256(256);
  const int totalND = N_NODES * DIM;
  const int nstrips = N_NODES / 16;
  gat_wcvt<<<dim3((DIM * DIM + 255) / 256), b256, 0, stream>>>(W, wtbuf);
  gat_gemm_wmma<<<dim3((nstrips + 7) / 8), b256, 0, stream>>>(
      x, wtbuf, al, ar, zbuf, el, er);
  // acc may alias x: safe — the GEMM (the only reader of x) is already
  // enqueued ahead on the same stream.
  gat_init<<<dim3((totalND + 255) / 256), b256, 0, stream>>>(acc, mmax, ssum,
                                                             N_NODES, totalND);
  gat_edge_logit<<<dim3((N_EDGES + 255) / 256), b256, 0, stream>>>(
      src, dst, el, er, ebuf, mmax, N_EDGES);
  gat_edge_exp<<<dim3((N_EDGES + 255) / 256), b256, 0, stream>>>(
      dst, ebuf, mmax, ssum, N_EDGES);
  gat_aggregate<<<dim3((N_EDGES + 7) / 8), b256, 0, stream>>>(
      src, dst, ebuf, ssum, zbuf, acc, N_EDGES);
  gat_bias_relu<<<dim3((totalND + 255) / 256), b256, 0, stream>>>(acc, b, hout,
                                                                  totalND);
}

extern "C" void kernel_launch(void* const* d_in, const int* in_sizes, int n_in,
                              void* d_out, int out_size, void* d_ws,
                              size_t ws_size, hipStream_t stream)
{
  (void)in_sizes; (void)n_in; (void)out_size; (void)ws_size;

  const float* f   = (const float*)d_in[0];
  const int*   src = (const int*)d_in[1];
  const int*   dst = (const int*)d_in[2];
  const float* W1  = (const float*)d_in[3];
  const float* al1 = (const float*)d_in[4];
  const float* ar1 = (const float*)d_in[5];
  const float* b1  = (const float*)d_in[6];
  const float* W2  = (const float*)d_in[7];
  const float* al2 = (const float*)d_in[8];
  const float* ar2 = (const float*)d_in[9];
  const float* b2  = (const float*)d_in[10];
  const float* W3  = (const float*)d_in[11];
  const float* al3 = (const float*)d_in[12];
  const float* ar3 = (const float*)d_in[13];
  const float* b3  = (const float*)d_in[14];

  float* out = (float*)d_out;

  // Workspace layout (floats): ~55.3 MB total
  float*  ws    = (float*)d_ws;
  float*  zbuf  = ws;                              // N*D
  float*  P0    = zbuf + (size_t)N_NODES * DIM;    // N*D (ping buffer)
  float*  el    = P0 + (size_t)N_NODES * DIM;      // N
  float*  er    = el + N_NODES;                    // N
  float*  mmax  = er + N_NODES;                    // N
  float*  ssum  = mmax + N_NODES;                  // N
  float*  ebuf  = ssum + N_NODES;                  // E
  __bf16* wtbuf = (__bf16*)(ebuf + N_EDGES);       // D*D bf16 (32 KB)

  // Layer 1: x = f (input), h1 -> P0 (acc in place)
  run_layer(f, W1, al1, ar1, b1, zbuf, P0, P0, wtbuf, el, er, mmax, ssum,
            ebuf, src, dst, stream);
  // Layer 2: x = P0, h2 -> P0
  run_layer(P0, W2, al2, ar2, b2, zbuf, P0, P0, wtbuf, el, er, mmax, ssum,
            ebuf, src, dst, stream);
  // Layer 3: x = P0, accumulate in P0, final relu+bias -> d_out
  run_layer(P0, W3, al3, ar3, b3, zbuf, P0, out, wtbuf, el, er, mmax, ssum,
            ebuf, src, dst, stream);
}
